// Head_58394375356491
// MI455X (gfx1250) — compile-verified, matrix operand order checked by reading, not run on hardware
//
#include <hip/hip_runtime.h>

// Problem constants (match reference)
#define BB 8
#define TT 2048
#define CC 1024
#define HH 64

typedef __attribute__((ext_vector_type(16))) __bf16 v16bf;
typedef __attribute__((ext_vector_type(8)))  __bf16 v8bf;
typedef __attribute__((ext_vector_type(8)))  float  v8f;
typedef __attribute__((ext_vector_type(4)))  int    v4i;

#define GAS __attribute__((address_space(1)))
#define LAS __attribute__((address_space(3)))

// ---- gfx1250 async global->LDS copy (ASYNCcnt-tracked), with sync fallback ----
#if defined(__has_builtin)
# if __has_builtin(__builtin_amdgcn_global_load_async_to_lds_b128)
#  define HAVE_ASYNC_LDS 1
# endif
#endif
#ifndef HAVE_ASYNC_LDS
# define HAVE_ASYNC_LDS 0
#endif

static __device__ __forceinline__ void copy_b128_to_lds(const __bf16* g, __bf16* l) {
#if HAVE_ASYNC_LDS
    __builtin_amdgcn_global_load_async_to_lds_b128(
        (GAS v4i*)(uintptr_t)g, (LAS v4i*)(uint32_t)(uintptr_t)l, 0, 0);
#else
    *(v8bf*)l = *(const v8bf*)g;
#endif
}

static __device__ __forceinline__ void async_wait0() {
#if HAVE_ASYNC_LDS
# if __has_builtin(__builtin_amdgcn_s_wait_asynccnt)
    __builtin_amdgcn_s_wait_asynccnt(0);
# else
    asm volatile("s_wait_asynccnt 0x0" ::: "memory");
# endif
#endif
}

static __device__ __forceinline__ v16bf frag_cat(v8bf lo, v8bf hi) {
    v16bf r;
#pragma unroll
    for (int i = 0; i < 8; ++i) { r[i] = lo[i]; r[i + 8] = hi[i]; }
    return r;
}

// ---------------------------------------------------------------------------
// Kernel 0 (one-time prep): W[C,64] f32 -> transposed bf16 [3][64][C].
// Makes every WMMA B-fragment of the projection GEMM two contiguous b128 loads.
// ---------------------------------------------------------------------------
__global__ __launch_bounds__(256) void prep_w_kernel(
    const float* __restrict__ Wq, const float* __restrict__ Wk,
    const float* __restrict__ Wv, __bf16* __restrict__ wTg)
{
    const int idx  = blockIdx.x * 256 + threadIdx.x;   // 0 .. 3*64*1024-1
    const int wsel = idx >> 16;
    const int rem  = idx & 65535;
    const int n    = rem >> 10;    // head dim 0..63
    const int c    = rem & 1023;   // embed dim 0..1023
    const float* W = (wsel == 0) ? Wq : ((wsel == 1) ? Wk : Wv);
    wTg[idx] = (__bf16)W[c * HH + n];
}

// ---------------------------------------------------------------------------
// Kernel 1: Q/K/V projection GEMM  [B,T,C] x [C,64] -> bf16 (x3).
// No LDS, no barriers: A-fragments straight from x, B-fragments from the
// pre-transposed bf16 weights (L2-resident). V stored tile-transposed
// ([tile][h][s_local]) so attention V-tiles are contiguous 4KB copies.
// ---------------------------------------------------------------------------
__global__ __launch_bounds__(128) void qkv_proj_kernel(
    const float* __restrict__ x,
    const __bf16* __restrict__ wTg,
    __bf16* __restrict__ Qw, __bf16* __restrict__ Kw, __bf16* __restrict__ Vt)
{
    const int tid    = threadIdx.x;
    const int wave   = tid >> 5;
    const int lane   = tid & 31;
    const int m      = lane & 15;
    const int halfid = lane >> 4;       // 0: lanes 0-15, 1: lanes 16-31
    const int k0     = halfid * 8;      // 16-bit A/B fragment K base (ISA layout)

    const int b  = blockIdx.x / (TT / 64);
    const int t0 = (blockIdx.x % (TT / 64)) * 64;
    const int myrow = t0 + wave * 16 + m;

    const float* xrow = x + (size_t)(b * TT + myrow) * CC;

    v8f acc[3][4];
#pragma unroll
    for (int wsel = 0; wsel < 3; ++wsel)
#pragma unroll
        for (int nt = 0; nt < 4; ++nt)
#pragma unroll
            for (int g = 0; g < 8; ++g) acc[wsel][nt][g] = 0.0f;

    for (int c0 = 0; c0 < CC; c0 += 32) {
        // A fragment: 16 x-values for row m, K-chunk [c0, c0+32)
        const float* xp = xrow + c0 + k0;
        float4 f0 = *(const float4*)(xp);
        float4 f1 = *(const float4*)(xp + 4);
        float4 f2 = *(const float4*)(xp + 16);
        float4 f3 = *(const float4*)(xp + 20);
        v16bf a;
        a[0]=(__bf16)f0.x; a[1]=(__bf16)f0.y; a[2]=(__bf16)f0.z; a[3]=(__bf16)f0.w;
        a[4]=(__bf16)f1.x; a[5]=(__bf16)f1.y; a[6]=(__bf16)f1.z; a[7]=(__bf16)f1.w;
        a[8]=(__bf16)f2.x; a[9]=(__bf16)f2.y; a[10]=(__bf16)f2.z; a[11]=(__bf16)f2.w;
        a[12]=(__bf16)f3.x; a[13]=(__bf16)f3.y; a[14]=(__bf16)f3.z; a[15]=(__bf16)f3.w;

#pragma unroll
        for (int wsel = 0; wsel < 3; ++wsel) {
#pragma unroll
            for (int nt = 0; nt < 4; ++nt) {
                const __bf16* bp =
                    wTg + (size_t)(wsel * HH + nt * 16 + m) * CC + c0 + k0;
                v16bf bm = frag_cat(*(const v8bf*)bp, *(const v8bf*)(bp + 16));
                acc[wsel][nt] = __builtin_amdgcn_wmma_f32_16x16x32_bf16(
                    false, a, false, bm, (short)0, acc[wsel][nt], false, false);
            }
        }
    }

#pragma unroll
    for (int nt = 0; nt < 4; ++nt)
#pragma unroll
        for (int g = 0; g < 8; ++g) {
            const int r = t0 + wave * 16 + g + halfid * 8;   // t index in batch
            const int h = nt * 16 + m;                       // head dim
            Qw[(size_t)(b * TT + r) * HH + h] = (__bf16)acc[0][nt][g];
            Kw[(size_t)(b * TT + r) * HH + h] = (__bf16)acc[1][nt][g];
            // V tile-transposed: [b][r/32][h][r%32]
            Vt[(size_t)b * TT * HH + (size_t)(r >> 5) * (HH * 32) + h * 32 + (r & 31)] =
                (__bf16)acc[2][nt][g];
        }
}

// ---------------------------------------------------------------------------
// Kernel 2: causal flash attention, bf16 WMMA + f32 online softmax.
// Double-buffered async K/V tile staging (both tiles are contiguous 4KB).
// ---------------------------------------------------------------------------
static __device__ __forceinline__ void stage_tile(
    const __bf16* Kbase, const __bf16* Vtbase, int tile,
    __bf16* ksd, __bf16* vtd, int tid)
{
    const __bf16* kg = Kbase  + (size_t)tile * (32 * HH) + tid * 8;
    const __bf16* vg = Vtbase + (size_t)tile * (32 * HH) + tid * 8;
    copy_b128_to_lds(kg,        ksd + tid * 8);
    copy_b128_to_lds(kg + 1024, ksd + tid * 8 + 1024);
    copy_b128_to_lds(vg,        vtd + tid * 8);
    copy_b128_to_lds(vg + 1024, vtd + tid * 8 + 1024);
}

__global__ __launch_bounds__(128) void flash_attn_kernel(
    const __bf16* __restrict__ Qw,
    const __bf16* __restrict__ Kw,
    const __bf16* __restrict__ Vt,
    float* __restrict__ out)
{
    __shared__ __align__(16) __bf16 ksb[2][32 * HH];    // row-major K tiles
    __shared__ __align__(16) __bf16 vTb[2][HH * 32];    // transposed V tiles
    __shared__ __align__(16) __bf16 ps[4][16 * 32];     // per-wave P staging

    const int tid    = threadIdx.x;
    const int wave   = tid >> 5;
    const int lane   = tid & 31;
    const int m      = lane & 15;
    const int halfid = lane >> 4;
    const int k0     = halfid * 8;

    const int b  = blockIdx.x / (TT / 64);
    const int q0 = (blockIdx.x % (TT / 64)) * 64;
    const int qw = q0 + wave * 16;

    const __bf16* Kbase  = Kw + (size_t)b * TT * HH;
    const __bf16* Vtbase = Vt + (size_t)b * TT * HH;

    // Q A-fragments (two 32-deep K chunks of the 64 head dims), loaded once
    const __bf16* qrow = Qw + (size_t)(b * TT + qw + m) * HH;
    v16bf aq[2];
#pragma unroll
    for (int c = 0; c < 2; ++c) {
        v8bf lo = *(const v8bf*)(qrow + c * 32 + k0);
        v8bf hi = *(const v8bf*)(qrow + c * 32 + k0 + 16);
        aq[c] = frag_cat(lo, hi);
    }

    v8f o[4];
    float rm[8], rs[8];
#pragma unroll
    for (int nt = 0; nt < 4; ++nt)
#pragma unroll
        for (int g = 0; g < 8; ++g) o[nt][g] = 0.0f;
#pragma unroll
    for (int g = 0; g < 8; ++g) { rm[g] = -1.0e30f; rs[g] = 0.0f; }

    const float kScale = 0.03125f;       // C^-0.5 = 1/32
    const float kL2E   = 1.44269504f;
    const int ntiles = q0 / 32 + 2;      // uniform across block (causal)

    stage_tile(Kbase, Vtbase, 0, ksb[0], vTb[0], tid);

    for (int st = 0; st < ntiles; ++st) {
        const int s0  = st * 32;
        const int buf = st & 1;

        async_wait0();
        __syncthreads();                 // tile `st` visible to all waves

        if (st + 1 < ntiles)
            stage_tile(Kbase, Vtbase, st + 1, ksb[buf ^ 1], vTb[buf ^ 1], tid);
        if (st + 2 < ntiles) {
            __builtin_prefetch(Kbase  + (size_t)(st + 2) * (32 * HH) + tid * 16, 0, 1);
            __builtin_prefetch(Vtbase + (size_t)(st + 2) * (32 * HH) + tid * 16, 0, 1);
        }

        // S = Q K^T : two 16x16 N-tiles, each accumulated over two K-chunks
        v8f sacc[2];
#pragma unroll
        for (int nt = 0; nt < 2; ++nt) {
#pragma unroll
            for (int g = 0; g < 8; ++g) sacc[nt][g] = 0.0f;
#pragma unroll
            for (int c = 0; c < 2; ++c) {
                const __bf16* bp = &ksb[buf][(nt * 16 + m) * HH + c * 32 + k0];
                v16bf bm = frag_cat(*(const v8bf*)bp, *(const v8bf*)(bp + 16));
                sacc[nt] = __builtin_amdgcn_wmma_f32_16x16x32_bf16(
                    false, aq[c], false, bm, (short)0, sacc[nt], false, false);
            }
        }

        // Online softmax in the WMMA C-layout: row M = g + 8*halfid, col N = m
        float p0[8], p1[8];
#pragma unroll
        for (int g = 0; g < 8; ++g) {
            const int qrowi = qw + g + halfid * 8;
            float v0 = sacc[0][g] * kScale;
            float v1 = sacc[1][g] * kScale;
            if (s0 + m      > qrowi) v0 = -1.0e30f;
            if (s0 + 16 + m > qrowi) v1 = -1.0e30f;
            float vm = fmaxf(v0, v1);
            vm = fmaxf(vm, __shfl_xor(vm, 8, 16));
            vm = fmaxf(vm, __shfl_xor(vm, 4, 16));
            vm = fmaxf(vm, __shfl_xor(vm, 2, 16));
            vm = fmaxf(vm, __shfl_xor(vm, 1, 16));
            const float mnew  = fmaxf(rm[g], vm);
            const float alpha = exp2f((rm[g] - mnew) * kL2E);
            p0[g] = exp2f((v0 - mnew) * kL2E);
            p1[g] = exp2f((v1 - mnew) * kL2E);
            float srow = p0[g] + p1[g];
            srow += __shfl_xor(srow, 8, 16);
            srow += __shfl_xor(srow, 4, 16);
            srow += __shfl_xor(srow, 2, 16);
            srow += __shfl_xor(srow, 1, 16);
            rs[g] = rs[g] * alpha + srow;
            rm[g] = mnew;
#pragma unroll
            for (int nt = 0; nt < 4; ++nt) o[nt][g] *= alpha;
        }

        // C-layout -> A-layout via per-wave LDS round trip
#pragma unroll
        for (int g = 0; g < 8; ++g) {
            const int r = g + halfid * 8;
            ps[wave][r * 32 + m]      = (__bf16)p0[g];
            ps[wave][r * 32 + 16 + m] = (__bf16)p1[g];
        }
        asm volatile("s_wait_dscnt 0" ::: "memory");

        const __bf16* pp = &ps[wave][m * 32 + k0];
        v16bf pa = frag_cat(*(const v8bf*)pp, *(const v8bf*)(pp + 16));

        // O += P V : four 16-wide N-tiles of the head dim
#pragma unroll
        for (int nt = 0; nt < 4; ++nt) {
            const __bf16* vp = &vTb[buf][(nt * 16 + m) * 32 + k0];
            v16bf bv = frag_cat(*(const v8bf*)vp, *(const v8bf*)(vp + 16));
            o[nt] = __builtin_amdgcn_wmma_f32_16x16x32_bf16(
                false, pa, false, bv, (short)0, o[nt], false, false);
        }
    }

    // Epilogue: normalize and store fp32 output
#pragma unroll
    for (int nt = 0; nt < 4; ++nt)
#pragma unroll
        for (int g = 0; g < 8; ++g) {
            const int r = qw + g + halfid * 8;
            out[(size_t)(b * TT + r) * HH + nt * 16 + m] = o[nt][g] / rs[g];
        }
}

// ---------------------------------------------------------------------------
extern "C" void kernel_launch(void* const* d_in, const int* in_sizes, int n_in,
                              void* d_out, int out_size, void* d_ws, size_t ws_size,
                              hipStream_t stream) {
    const float* x  = (const float*)d_in[0];
    // d_in[1] = mask (int32) — causal mask applied analytically, unused
    const float* Wq = (const float*)d_in[2];
    const float* Wk = (const float*)d_in[3];
    const float* Wv = (const float*)d_in[4];
    float* out = (float*)d_out;

    const size_t nQKV = (size_t)BB * TT * HH;           // 1,048,576 elements
    __bf16* Qw  = (__bf16*)d_ws;
    __bf16* Kw  = Qw + nQKV;
    __bf16* Vt  = Kw + nQKV;
    __bf16* wTg = Vt + nQKV;                            // 3*64*1024 bf16

    prep_w_kernel<<<(3 * HH * CC) / 256, 256, 0, stream>>>(Wq, Wk, Wv, wTg);

    const int nblk = (BB * TT) / 64;                    // 256
    qkv_proj_kernel<<<nblk, 128, 0, stream>>>(x, wTg, Qw, Kw, Vt);
    flash_attn_kernel<<<nblk, 128, 0, stream>>>(Qw, Kw, Vt, out);
}
